// MoEPreActivationResBlock_9560597201203
// MI455X (gfx1250) — matrix operand order repaired and checked by reading, not compiled, not run on hardware
//
#include <hip/hip_runtime.h>

#define N_TOK 16384
#define DDIM  1024
#define HDIM  2048
#define NEXP  8
#define TOPK  2
#define CAPN  4096
#define EPSF  1e-6f

typedef __attribute__((ext_vector_type(16))) _Float16 v16h;
typedef __attribute__((ext_vector_type(8)))  _Float16 v8h;
typedef __attribute__((ext_vector_type(8)))  float    v8f;

union V16 { v16h v; v8h h[2]; };

// --- CDNA5 async global->LDS path (ASYNCcnt), with compile-safe fallback ----
#if defined(__gfx1250__) &&                                                    \
    __has_builtin(__builtin_amdgcn_global_load_async_to_lds_b128) &&           \
    __has_builtin(__builtin_amdgcn_s_wait_asynccnt)
#define HAVE_ASYNC_LDS 1
#else
#define HAVE_ASYNC_LDS 0
#endif

#if HAVE_ASYNC_LDS
typedef int v4i_ __attribute__((vector_size(4 * sizeof(int))));
typedef __attribute__((address_space(1))) v4i_ as1_v4i;
typedef __attribute__((address_space(3))) v4i_ as3_v4i;

static __device__ __forceinline__ void async_b128(void* lds_dst,
                                                  const void* gsrc) {
  __builtin_amdgcn_global_load_async_to_lds_b128(
      (as1_v4i*)gsrc, (as3_v4i*)lds_dst, /*imm offset*/ 0, /*cpol*/ 0);
}
#endif

static __device__ __forceinline__ unsigned int ballot32(bool p) {
#if __has_builtin(__builtin_amdgcn_ballot_w32)
  return __builtin_amdgcn_ballot_w32(p);
#else
  return (unsigned int)__ballot(p ? 1 : 0);
#endif
}

// ---------------------------------------------------------------------------
// Prep: per-expert transpose + f32->f16 convert.  in: [E][R][C] f32,
// out: [E][C][R] f16 (so WMMA B fragments are contiguous along K).
// ---------------------------------------------------------------------------
__global__ void k_transpose_f16(const float* __restrict__ in,
                                _Float16* __restrict__ out, int R, int C) {
  const size_t rc = (size_t)R * C;
  const size_t total = (size_t)NEXP * rc;
  for (size_t idx = (size_t)blockIdx.x * blockDim.x + threadIdx.x; idx < total;
       idx += (size_t)gridDim.x * blockDim.x) {
    const size_t e = idx / rc, rem = idx % rc;
    const size_t c = rem / (size_t)R, r = rem % (size_t)R;
    out[idx] = (_Float16)in[(e * (size_t)R + r) * (size_t)C + c];
  }
}

__global__ void k_copy_f32v4(const float4* __restrict__ src,
                             float4* __restrict__ dst, size_t n4) {
  for (size_t i = (size_t)blockIdx.x * blockDim.x + threadIdx.x; i < n4;
       i += (size_t)gridDim.x * blockDim.x)
    dst[i] = src[i];
}

// ---------------------------------------------------------------------------
// Fused LayerNorm + ReLU + router logits.  One 256-thread block per token.
// ---------------------------------------------------------------------------
__global__ __launch_bounds__(256) void k_ln_relu_router(
    const float* __restrict__ x0, const float* __restrict__ ln_scale,
    const float* __restrict__ ln_bias, const float* __restrict__ Wr,
    const float* __restrict__ br, _Float16* __restrict__ xnh,
    float* __restrict__ logits) {
  const int t = blockIdx.x;
  const int tid = threadIdx.x;
  const int lane = tid & 31, wave = tid >> 5;  // 8 waves of 32 (wave32)

  const float4 xv = ((const float4*)(x0 + (size_t)t * DDIM))[tid];
  float xr[4] = {xv.x, xv.y, xv.z, xv.w};

  float s = xr[0] + xr[1] + xr[2] + xr[3];
  float ss = xr[0]*xr[0] + xr[1]*xr[1] + xr[2]*xr[2] + xr[3]*xr[3];
#pragma unroll
  for (int off = 16; off > 0; off >>= 1) {
    s  += __shfl_xor(s,  off, 32);
    ss += __shfl_xor(ss, off, 32);
  }
  __shared__ float rs[8], rss[8];
  if (lane == 0) { rs[wave] = s; rss[wave] = ss; }
  __syncthreads();
  float st = 0.f, sst = 0.f;
#pragma unroll
  for (int w = 0; w < 8; ++w) { st += rs[w]; sst += rss[w]; }
  const float mu   = st  * (1.f / DDIM);
  const float var  = sst * (1.f / DDIM) - mu * mu;
  const float rstd = rsqrtf(var + EPSF);

  const int d0 = tid * 4;
  float xn[4];
#pragma unroll
  for (int c = 0; c < 4; ++c) {
    float v = (xr[c] - mu) * rstd * ln_scale[d0 + c] + ln_bias[d0 + c];
    xn[c] = fmaxf(v, 0.f);
  }
  _Float16* orow = xnh + (size_t)t * DDIM + d0;
#pragma unroll
  for (int c = 0; c < 4; ++c) orow[c] = (_Float16)xn[c];

  // Router: logits[t][e] = xn . Wr[:, e]   (Wr is [D][E] row-major, E=8)
  float lg[8] = {0, 0, 0, 0, 0, 0, 0, 0};
#pragma unroll
  for (int c = 0; c < 4; ++c) {
    const float* wr = Wr + (size_t)(d0 + c) * NEXP;
    const float4 w0 = ((const float4*)wr)[0];
    const float4 w1 = ((const float4*)wr)[1];
    lg[0] += xn[c] * w0.x; lg[1] += xn[c] * w0.y;
    lg[2] += xn[c] * w0.z; lg[3] += xn[c] * w0.w;
    lg[4] += xn[c] * w1.x; lg[5] += xn[c] * w1.y;
    lg[6] += xn[c] * w1.z; lg[7] += xn[c] * w1.w;
  }
#pragma unroll
  for (int e = 0; e < 8; ++e) {
#pragma unroll
    for (int off = 16; off > 0; off >>= 1) lg[e] += __shfl_xor(lg[e], off, 32);
  }
  __shared__ float racc[8][8];
  if (lane == 0) {
#pragma unroll
    for (int e = 0; e < 8; ++e) racc[wave][e] = lg[e];
  }
  __syncthreads();
  if (tid < 8) {
    float a = br[tid];
#pragma unroll
    for (int w = 0; w < 8; ++w) a += racc[w][tid];
    logits[(size_t)t * NEXP + tid] = a;
  }
}

// ---------------------------------------------------------------------------
// Top-2 + softmax gates per token.
// ---------------------------------------------------------------------------
__global__ void k_top2(const float* __restrict__ logits,
                       int* __restrict__ eidx, float* __restrict__ gatef) {
  const int t = blockIdx.x * blockDim.x + threadIdx.x;
  if (t >= N_TOK) return;
  float l[8];
#pragma unroll
  for (int e = 0; e < 8; ++e) l[e] = logits[(size_t)t * NEXP + e];
  int i0 = 0; float v0 = l[0];
#pragma unroll
  for (int e = 1; e < 8; ++e) if (l[e] > v0) { v0 = l[e]; i0 = e; }
  int i1 = (i0 == 0) ? 1 : 0; float v1 = l[i1];
#pragma unroll
  for (int e = 0; e < 8; ++e)
    if (e != i0 && l[e] > v1) { v1 = l[e]; i1 = e; }
  const float g0 = 1.f / (1.f + __expf(v1 - v0));
  eidx[2 * t]     = i0;  eidx[2 * t + 1]  = i1;
  gatef[2 * t]    = g0;  gatef[2 * t + 1] = 1.f - g0;
}

// ---------------------------------------------------------------------------
// Deterministic token-order capacity scan: one wave per expert, wave32 ballot
// prefix (matches cumsum(one_hot) - 1 of the reference exactly).
// ---------------------------------------------------------------------------
__global__ __launch_bounds__(256) void k_scan_pos(const int* __restrict__ eidx,
                                                  int* __restrict__ pos) {
  const int lane = threadIdx.x & 31;
  const int myexp = threadIdx.x >> 5;  // waves 0..7 <-> experts 0..7
  if (myexp >= NEXP) return;
  int cnt = 0;
  for (int base = 0; base < N_TOK * TOPK; base += 32) {
    const int i = base + lane;
    const bool match = (eidx[i] == myexp);
    const unsigned m = ballot32(match);
    if (match) pos[i] = cnt + __popc(m & ((1u << lane) - 1u));
    cnt += __popc(m);
  }
}

// ---------------------------------------------------------------------------
// Dispatch: copy kept slots into [E][CAP][D] f16 buffer, record inverse map.
// ---------------------------------------------------------------------------
__global__ __launch_bounds__(128) void k_scatter(
    const _Float16* __restrict__ xnh, const int* __restrict__ eidx,
    const int* __restrict__ pos, const float* __restrict__ gatef,
    _Float16* __restrict__ Xe, int* __restrict__ smap,
    float* __restrict__ sgate) {
  const int s = blockIdx.x;
  const int e = eidx[s], p = pos[s];
  if (p >= CAPN) return;  // over capacity -> dropped
  const int tok = s >> 1;
  const uint4* src = (const uint4*)(xnh + (size_t)tok * DDIM);
  uint4* dst = (uint4*)(Xe + ((size_t)e * CAPN + p) * DDIM);
  dst[threadIdx.x] = src[threadIdx.x];
  if (threadIdx.x == 0) {
    smap[e * CAPN + p]  = s;
    sgate[e * CAPN + p] = gatef[s];
  }
}

// ---------------------------------------------------------------------------
// GEMM1 + bias + LayerNorm(H) + ReLU, f16 WMMA with f32 accumulate.
// Block = 512 threads = 16 waves; 16 x 2048 row strip (LN spans all of H).
// Async double-buffered LDS staging on CDNA5 (ASYNCcnt pipeline).
// Per-wave async ops per stage: 1 (A row, lanes 0-3) + 16 (B) = 17.
// LDS: As0|As1 (1KB each) @0/@1024; Bs0|Bs1 (128KB each) @2048; Ht reuses Bs0.
// ---------------------------------------------------------------------------
__global__ __launch_bounds__(512) void k_gemm1_ln(
    const _Float16* __restrict__ Xe, const _Float16* __restrict__ W1t,
    const float* __restrict__ b1, const float* __restrict__ n_scale,
    const float* __restrict__ n_bias, _Float16* __restrict__ Hf) {
  const int e    = blockIdx.y;
  const int row0 = blockIdx.x * 16;
  const int tid  = threadIdx.x;
  const int lane = tid & 31, wave = tid >> 5;  // 16 waves
  const int m16  = lane & 15, kh = lane >> 4;
  const int wn0  = wave * 128;

  extern __shared__ char smem[];
  _Float16* const AsBase = (_Float16*)smem;           // 2 x [16][32]
  _Float16* const BsBase = (_Float16*)(smem + 2048);  // 2 x [2048][32]
  float* const Ht = (float*)(smem + 2048);            // [16][2048], reuses Bs0

  const _Float16* Ag = Xe + ((size_t)e * CAPN + row0) * DDIM;
  const _Float16* Bg = W1t + (size_t)e * HDIM * DDIM;  // [n][k], stride DDIM

  v8f acc[8] = {};

#if HAVE_ASYNC_LDS
  // -------- stage a K-step into buffer b (uniform 17 async ops / wave) -----
  auto stage = [&](int kk, int b) {
    _Float16* Asd = AsBase + b * 512;          // 16*32 f16 per buffer
    _Float16* Bsd = BsBase + b * (HDIM * 32);  // 2048*32 f16 per buffer
    if (lane < 4)  // every wave stages one A row -> exactly 1 instr per wave
      async_b128(Asd + wave * 32 + lane * 8,
                 Ag + (size_t)wave * DDIM + kk + lane * 8);
    for (int i = tid; i < HDIM * 4; i += 512) {  // 16 instrs per thread
      const int n = i >> 2, c4 = i & 3;
      async_b128(Bsd + n * 32 + c4 * 8,
                 Bg + (size_t)n * DDIM + kk + c4 * 8);
    }
  };
  stage(0, 0);
  for (int k0 = 0, it = 0; k0 < DDIM; k0 += 32, ++it) {
    const int cur = it & 1;
    if (k0 + 32 < DDIM) {
      stage(k0 + 32, cur ^ 1);                   // overlap next stage
      __builtin_amdgcn_s_wait_asynccnt(17);      // current stage landed
    } else {
      __builtin_amdgcn_s_wait_asynccnt(0);
    }
    __syncthreads();                              // publish LDS across waves
    _Float16* Asc = AsBase + cur * 512;
    _Float16* Bsc = BsBase + cur * (HDIM * 32);
    V16 af;
    af.h[0] = *(const v8h*)(Asc + m16 * 32 + kh * 8);
    af.h[1] = *(const v8h*)(Asc + m16 * 32 + 16 + kh * 8);
#pragma unroll
    for (int t = 0; t < 8; ++t) {
      const int n = wn0 + t * 16 + m16;
      V16 bf;
      bf.h[0] = *(const v8h*)(Bsc + n * 32 + kh * 8);
      bf.h[1] = *(const v8h*)(Bsc + n * 32 + 16 + kh * 8);
      acc[t] = __builtin_amdgcn_wmma_f32_16x16x32_f16(
          false, af.v, false, bf.v, (short)0, acc[t], false, false);
    }
    __syncthreads();  // all waves done reading before buffer is re-staged
  }
#else
  // -------- fallback: synchronous staging through VGPRs --------------------
  _Float16* As = AsBase;
  _Float16* Bs = BsBase;
  for (int k0 = 0; k0 < DDIM; k0 += 32) {
    __syncthreads();
    if (tid < 64) {
      const int r = tid >> 2, c4 = tid & 3;
      *(uint4*)(As + r * 32 + c4 * 8) =
          *(const uint4*)(Ag + (size_t)r * DDIM + k0 + c4 * 8);
    }
    for (int i = tid; i < HDIM * 4; i += 512) {
      const int n = i >> 2, c4 = i & 3;
      *(uint4*)(Bs + n * 32 + c4 * 8) =
          *(const uint4*)(Bg + (size_t)n * DDIM + k0 + c4 * 8);
    }
    if (k0 + 32 < DDIM) __builtin_prefetch(Bg + k0 + 32, 0, 0);
    __syncthreads();
    V16 af;
    af.h[0] = *(const v8h*)(As + m16 * 32 + kh * 8);
    af.h[1] = *(const v8h*)(As + m16 * 32 + 16 + kh * 8);
#pragma unroll
    for (int t = 0; t < 8; ++t) {
      const int n = wn0 + t * 16 + m16;
      V16 bf;
      bf.h[0] = *(const v8h*)(Bs + n * 32 + kh * 8);
      bf.h[1] = *(const v8h*)(Bs + n * 32 + 16 + kh * 8);
      acc[t] = __builtin_amdgcn_wmma_f32_16x16x32_f16(
          false, af.v, false, bf.v, (short)0, acc[t], false, false);
    }
  }
  __syncthreads();
#endif

  // Spill accumulators (+b1) to LDS so LayerNorm can span the full row.
#pragma unroll
  for (int t = 0; t < 8; ++t) {
    const int n = wn0 + t * 16 + m16;
    const float bb = b1[e * HDIM + n];
#pragma unroll
    for (int j = 0; j < 8; ++j) {
      const int m = kh * 8 + j;  // C layout: VGPR j -> M=j (lanes<16) / 8+j
      Ht[m * HDIM + n] = acc[t][j] + bb;
    }
  }
  __syncthreads();

  // LayerNorm + ReLU: wave w normalizes row w (16 waves / 16 rows).
  {
    const int m = wave;
    const int nbase = lane * 64;
    float s = 0.f, ss = 0.f;
    for (int i = 0; i < 64; ++i) {
      const float v = Ht[m * HDIM + nbase + i];
      s += v; ss += v * v;
    }
#pragma unroll
    for (int off = 16; off > 0; off >>= 1) {
      s += __shfl_xor(s, off, 32); ss += __shfl_xor(ss, off, 32);
    }
    const float mu   = s * (1.f / HDIM);
    const float var  = ss * (1.f / HDIM) - mu * mu;
    const float rstd = rsqrtf(var + EPSF);
    _Float16* orow = Hf + ((size_t)e * CAPN + row0 + m) * HDIM;
    for (int i = 0; i < 64; ++i) {
      const int n = nbase + i;
      float v = (Ht[m * HDIM + n] - mu) * rstd * n_scale[e * HDIM + n] +
                n_bias[e * HDIM + n];
      orow[n] = (_Float16)fmaxf(v, 0.f);
    }
  }
}

// ---------------------------------------------------------------------------
// GEMM2 + bias + gate + atomic combine into residual-initialized output.
// Block = 512 threads = 16 waves; 16 rows x 1024 cols, K = 2048.
// Per-wave async ops per stage: 1 (A) + 8 (B) = 9.
// ---------------------------------------------------------------------------
__global__ __launch_bounds__(512) void k_gemm2_combine(
    const _Float16* __restrict__ Hf, const _Float16* __restrict__ W2t,
    const float* __restrict__ b2, const int* __restrict__ smap,
    const float* __restrict__ sgate, float* __restrict__ out) {
  const int e    = blockIdx.y;
  const int row0 = blockIdx.x * 16;
  const int tid  = threadIdx.x;
  const int lane = tid & 31, wave = tid >> 5;
  const int m16  = lane & 15, kh = lane >> 4;
  const int wn0  = wave * 64;  // 16 waves * 64 = 1024 output columns

  extern __shared__ char smem[];
  _Float16* const AsBase = (_Float16*)smem;           // 2 x [16][32]
  _Float16* const BsBase = (_Float16*)(smem + 2048);  // 2 x [1024][32]
  int*   const Sm = (int*)(smem + 2048 + 2 * (size_t)DDIM * 32 * 2);
  float* const Sg = (float*)(smem + 2048 + 2 * (size_t)DDIM * 32 * 2 + 64);

  if (tid < 16) {
    Sm[tid] = smap[e * CAPN + row0 + tid];
    Sg[tid] = sgate[e * CAPN + row0 + tid];
  }

  const _Float16* Ag = Hf + ((size_t)e * CAPN + row0) * HDIM;
  const _Float16* Bg = W2t + (size_t)e * DDIM * HDIM;  // [n][k], stride HDIM

  v8f acc[4] = {};

#if HAVE_ASYNC_LDS
  auto stage = [&](int kk, int b) {
    _Float16* Asd = AsBase + b * 512;
    _Float16* Bsd = BsBase + b * (DDIM * 32);
    if (lane < 4)
      async_b128(Asd + wave * 32 + lane * 8,
                 Ag + (size_t)wave * HDIM + kk + lane * 8);
    for (int i = tid; i < DDIM * 4; i += 512) {  // 8 instrs per thread
      const int n = i >> 2, c4 = i & 3;
      async_b128(Bsd + n * 32 + c4 * 8,
                 Bg + (size_t)n * HDIM + kk + c4 * 8);
    }
  };
  stage(0, 0);
  for (int k0 = 0, it = 0; k0 < HDIM; k0 += 32, ++it) {
    const int cur = it & 1;
    if (k0 + 32 < HDIM) {
      stage(k0 + 32, cur ^ 1);
      __builtin_amdgcn_s_wait_asynccnt(9);
    } else {
      __builtin_amdgcn_s_wait_asynccnt(0);
    }
    __syncthreads();
    _Float16* Asc = AsBase + cur * 512;
    _Float16* Bsc = BsBase + cur * (DDIM * 32);
    V16 af;
    af.h[0] = *(const v8h*)(Asc + m16 * 32 + kh * 8);
    af.h[1] = *(const v8h*)(Asc + m16 * 32 + 16 + kh * 8);
#pragma unroll
    for (int t = 0; t < 4; ++t) {
      const int n = wn0 + t * 16 + m16;
      V16 bf;
      bf.h[0] = *(const v8h*)(Bsc + n * 32 + kh * 8);
      bf.h[1] = *(const v8h*)(Bsc + n * 32 + 16 + kh * 8);
      acc[t] = __builtin_amdgcn_wmma_f32_16x16x32_f16(
          false, af.v, false, bf.v, (short)0, acc[t], false, false);
    }
    __syncthreads();
  }
#else
  _Float16* As = AsBase;
  _Float16* Bs = BsBase;
  for (int k0 = 0; k0 < HDIM; k0 += 32) {
    __syncthreads();
    if (tid < 64) {
      const int r = tid >> 2, c4 = tid & 3;
      *(uint4*)(As + r * 32 + c4 * 8) =
          *(const uint4*)(Ag + (size_t)r * HDIM + k0 + c4 * 8);
    }
    for (int i = tid; i < DDIM * 4; i += 512) {
      const int n = i >> 2, c4 = i & 3;
      *(uint4*)(Bs + n * 32 + c4 * 8) =
          *(const uint4*)(Bg + (size_t)n * HDIM + k0 + c4 * 8);
    }
    if (k0 + 32 < HDIM) __builtin_prefetch(Bg + k0 + 32, 0, 0);
    __syncthreads();
    V16 af;
    af.h[0] = *(const v8h*)(As + m16 * 32 + kh * 8);
    af.h[1] = *(const v8h*)(As + m16 * 32 + 16 + kh * 8);
#pragma unroll
    for (int t = 0; t < 4; ++t) {
      const int n = wn0 + t * 16 + m16;
      V16 bf;
      bf.h[0] = *(const v8h*)(Bs + n * 32 + kh * 8);
      bf.h[1] = *(const v8h*)(Bs + n * 32 + 16 + kh * 8);
      acc[t] = __builtin_amdgcn_wmma_f32_16x16x32_f16(
          false, af.v, false, bf.v, (short)0, acc[t], false, false);
    }
  }
  __syncthreads();
#endif

#pragma unroll
  for (int t = 0; t < 4; ++t) {
    const int n = wn0 + t * 16 + m16;
    const float bb = b2[e * DDIM + n];
#pragma unroll
    for (int j = 0; j < 8; ++j) {
      const int m = kh * 8 + j;
      const int s = Sm[m];
      if (s >= 0) {
        const int tok = s >> 1;  // slot -> token (K=2)
        unsafeAtomicAdd(out + (size_t)tok * DDIM + n, (acc[t][j] + bb) * Sg[m]);
      }
    }
  }
}

// ---------------------------------------------------------------------------
extern "C" void kernel_launch(void* const* d_in, const int* in_sizes, int n_in,
                              void* d_out, int out_size, void* d_ws,
                              size_t ws_size, hipStream_t stream) {
  (void)in_sizes; (void)n_in; (void)out_size; (void)ws_size;
  const float* x0       = (const float*)d_in[0];
  const float* ln_scale = (const float*)d_in[1];
  const float* ln_bias  = (const float*)d_in[2];
  const float* Wr       = (const float*)d_in[3];
  const float* br       = (const float*)d_in[4];
  const float* W1       = (const float*)d_in[5];
  const float* b1       = (const float*)d_in[6];
  const float* n_scale  = (const float*)d_in[7];
  const float* n_bias   = (const float*)d_in[8];
  const float* W2       = (const float*)d_in[9];
  const float* b2       = (const float*)d_in[10];
  float* out = (float*)d_out;

  char* ws = (char*)d_ws;
  size_t off = 0;
  auto carve = [&](size_t bytes) -> void* {
    void* p = ws + off;
    off += (bytes + 255) & ~(size_t)255;
    return p;
  };
  _Float16* xnh   = (_Float16*)carve((size_t)N_TOK * DDIM * 2);       // 32 MB
  _Float16* W1t   = (_Float16*)carve((size_t)NEXP * HDIM * DDIM * 2); // 32 MB
  _Float16* W2t   = (_Float16*)carve((size_t)NEXP * DDIM * HDIM * 2); // 32 MB
  _Float16* Xe    = (_Float16*)carve((size_t)NEXP * CAPN * DDIM * 2); // 64 MB
  _Float16* Hf    = (_Float16*)carve((size_t)NEXP * CAPN * HDIM * 2); // 128 MB
  float*    logit = (float*)carve((size_t)N_TOK * NEXP * 4);
  int*      eidx  = (int*)carve((size_t)N_TOK * TOPK * 4);
  float*    gatef = (float*)carve((size_t)N_TOK * TOPK * 4);
  int*      pos   = (int*)carve((size_t)N_TOK * TOPK * 4);
  int*      smap  = (int*)carve((size_t)NEXP * CAPN * 4);
  float*    sgate = (float*)carve((size_t)NEXP * CAPN * 4);

  // 0) weight transpose/convert + residual init + buffer clears
  k_transpose_f16<<<4096, 256, 0, stream>>>(W1, W1t, DDIM, HDIM);
  k_transpose_f16<<<4096, 256, 0, stream>>>(W2, W2t, HDIM, DDIM);
  k_copy_f32v4<<<8192, 256, 0, stream>>>((const float4*)x0, (float4*)out,
                                         (size_t)N_TOK * DDIM / 4);
  (void)hipMemsetAsync(Xe, 0, (size_t)NEXP * CAPN * DDIM * 2, stream);
  (void)hipMemsetAsync(smap, 0xFF, (size_t)NEXP * CAPN * 4, stream);  // -1

  // 1) LN + ReLU + router logits
  k_ln_relu_router<<<N_TOK, 256, 0, stream>>>(x0, ln_scale, ln_bias, Wr, br,
                                              xnh, logit);
  // 2) top-2 gates
  k_top2<<<N_TOK / 256, 256, 0, stream>>>(logit, eidx, gatef);
  // 3) deterministic capacity scan (one wave per expert)
  k_scan_pos<<<1, 256, 0, stream>>>(eidx, pos);
  // 4) dispatch
  k_scatter<<<N_TOK * TOPK, 128, 0, stream>>>(xnh, eidx, pos, gatef, Xe, smap,
                                              sgate);
  // 5) expert GEMM1 + LN + ReLU  (WMMA, async double-buffered staging)
  {
    dim3 g(CAPN / 16, NEXP);
    const size_t sm = 2048 + 2 * (size_t)HDIM * 32 * 2;  // 258 KB
    k_gemm1_ln<<<g, 512, sm, stream>>>(Xe, W1t, b1, n_scale, n_bias, Hf);
  }
  // 6) expert GEMM2 + gated combine  (WMMA, async double-buffered staging)
  {
    dim3 g(CAPN / 16, NEXP);
    const size_t sm = 2048 + 2 * (size_t)DDIM * 32 * 2 + 128;  // 130 KB + maps
    k_gemm2_combine<<<g, 512, sm, stream>>>(Hf, W2t, b2, smap, sgate, out);
  }
}
